// MoeLoopBlock_17480516895115
// MI455X (gfx1250) — matrix operand order repaired and checked by reading, not compiled, not run on hardware
//
#include <hip/hip_runtime.h>
#include <hip/hip_bf16.h>

// Problem dims (fixed by the reference)
#define N_TOK 8192   // B*S = 4*2048
#define DDIM  1024
#define FDIM  4096
#define NEXP  8

// GEMM tiling
#define BM 128
#define BN 128
#define BK 32

typedef __attribute__((ext_vector_type(16))) __bf16 v16bf;
typedef __attribute__((ext_vector_type(8)))  float  v8f;

union FragU { uint4 u[2]; v16bf v; };

// packed fp32x2 -> bf16x2 via native v_cvt_pk_bf16_f32 (gfx1250 bf16 datapath)
__device__ __forceinline__ unsigned int pack2bf(float a, float b) {
  unsigned int r;
  asm("v_cvt_pk_bf16_f32 %0, %1, %2" : "=v"(r) : "v"(a), "v"(b));
  return r;
}
__device__ __forceinline__ unsigned short f2bf(float f) {
  return (unsigned short)pack2bf(f, 0.f);
}
__device__ __forceinline__ float f4c(const float4& v, int c) { return (&v.x)[c]; }

// ---- async global->LDS copy (CDNA5) ---------------------------------------
typedef int v4i __attribute__((vector_size(4 * sizeof(int))));
typedef __attribute__((address_space(1))) v4i gv4i;   // global
typedef __attribute__((address_space(3))) v4i lv4i;   // LDS

__device__ __forceinline__ void copy32_g2l_async(const void* gp, void* lp) {
#if __has_builtin(__builtin_amdgcn_global_load_async_to_lds_b128)
  __builtin_amdgcn_global_load_async_to_lds_b128((gv4i*)gp, (lv4i*)lp, 0, 0);
  __builtin_amdgcn_global_load_async_to_lds_b128((gv4i*)gp, (lv4i*)lp, 16, 0);
#else
  const uint4* s = (const uint4*)gp;
  uint4* d = (uint4*)lp;
  d[0] = s[0];
  d[1] = s[1];
#endif
}
__device__ __forceinline__ void wait_async0() {
#if __has_builtin(__builtin_amdgcn_global_load_async_to_lds_b128)
#if __has_builtin(__builtin_amdgcn_s_wait_asynccnt)
  __builtin_amdgcn_s_wait_asynccnt(0);
#else
  asm volatile("s_wait_asynccnt 0x0" ::: "memory");
#endif
#endif
}

// ---------------------------------------------------------------------------
// Gating: one wave32 per token. logits = x_row @ wg, top-2 + softmax,
// dense per-expert weight vector (0 for unselected experts).
// ---------------------------------------------------------------------------
__global__ void gate_kernel(const float* __restrict__ x,
                            const float* __restrict__ wg,
                            float* __restrict__ we) {
  const int lane = threadIdx.x & 31;
  const int wv   = threadIdx.x >> 5;
  const int tok  = blockIdx.x * 8 + wv;
  const float* xr = x + (size_t)tok * DDIM;

  float s[NEXP];
#pragma unroll
  for (int e = 0; e < NEXP; ++e) s[e] = 0.f;

  for (int d = lane; d < DDIM; d += 32) {
    float xv = xr[d];
#pragma unroll
    for (int e = 0; e < NEXP; ++e) s[e] += xv * wg[d * NEXP + e];
  }
#pragma unroll
  for (int off = 16; off > 0; off >>= 1) {
#pragma unroll
    for (int e = 0; e < NEXP; ++e) s[e] += __shfl_xor(s[e], off, 32);
  }
  if (lane == 0) {
    int i1 = 0;
#pragma unroll
    for (int e = 1; e < NEXP; ++e) if (s[e] > s[i1]) i1 = e;
    int i2 = (i1 == 0) ? 1 : 0;
#pragma unroll
    for (int e = 0; e < NEXP; ++e) if (e != i1 && s[e] > s[i2]) i2 = e;
    float p1 = 1.f / (1.f + __expf(s[i2] - s[i1]));   // softmax over top-2
    float p2 = 1.f - p1;
#pragma unroll
    for (int e = 0; e < NEXP; ++e)
      we[(size_t)tok * NEXP + e] = (e == i1) ? p1 : ((e == i2) ? p2 : 0.f);
  }
}

// ---------------------------------------------------------------------------
// Stage 1 per expert: Hb[N,F] = silu(x @ w0e) * (x @ w1e), bf16 output.
// fp32 inputs converted to bf16 while staging into LDS; bf16 WMMA MACs.
// Block = 256 threads (8 waves, 2x4 over the 128x128 tile); wave tile 64x32.
// ---------------------------------------------------------------------------
__global__ __launch_bounds__(256, 1)
void moe_h_kernel(const float* __restrict__ x,
                  const float* __restrict__ w0,   // [D][F] this expert
                  const float* __restrict__ w1,   // [D][F] this expert
                  unsigned short* __restrict__ Hb) {
  __shared__ __align__(16) unsigned short ldsA [BM * BK];  // [m][k]
  __shared__ __align__(16) unsigned short ldsB0[BN * BK];  // [n][k] (transposed)
  __shared__ __align__(16) unsigned short ldsB1[BN * BK];

  const int tid  = threadIdx.x;
  const int lane = tid & 31;
  const int wv   = tid >> 5;
  const int wm   = wv >> 2;            // 0..1 -> 64 rows
  const int wn   = wv & 3;             // 0..3 -> 32 cols
  const int bn0  = blockIdx.x * BN;
  const int bm0  = blockIdx.y * BM;

  // A staging: thread -> (row, 16-float half of the 32-wide k slice)
  const int ar = tid >> 1;
  const int ah = (tid & 1) * 16;
  // B staging: thread -> 4(k) x 4(n) sub-tile, packed ds_store_b64 writes
  const int bkq = (tid >> 5) * 4;      // 0,4,..,28
  const int bnn = (tid & 31) * 4;      // 0,4,..,124

  v8f acc0[4][2], acc1[4][2];
#pragma unroll
  for (int i = 0; i < 4; ++i)
#pragma unroll
    for (int j = 0; j < 2; ++j)
#pragma unroll
      for (int r = 0; r < 8; ++r) { acc0[i][j][r] = 0.f; acc1[i][j][r] = 0.f; }

  const int hi = lane >> 4;
  const int lm = lane & 15;

  for (int kk = 0; kk < DDIM; kk += BK) {
    // ---- stage A (x fp32 -> bf16, packed uint4 stores) ----
    {
      const float4* ap = (const float4*)(x + (size_t)(bm0 + ar) * DDIM + kk + ah);
      float4 a0 = ap[0], a1 = ap[1], a2 = ap[2], a3 = ap[3];
      uint4* dst = (uint4*)&ldsA[ar * BK + ah];
      dst[0] = make_uint4(pack2bf(a0.x, a0.y), pack2bf(a0.z, a0.w),
                          pack2bf(a1.x, a1.y), pack2bf(a1.z, a1.w));
      dst[1] = make_uint4(pack2bf(a2.x, a2.y), pack2bf(a2.z, a2.w),
                          pack2bf(a3.x, a3.y), pack2bf(a3.z, a3.w));
    }
    // ---- stage B0/B1 (w fp32 -> bf16, transposed [n][k], b64 stores) ----
    {
      const float* b0r = w0 + (size_t)(kk + bkq) * FDIM + bn0 + bnn;
      const float* b1r = w1 + (size_t)(kk + bkq) * FDIM + bn0 + bnn;
      float4 r0[4], r1[4];
#pragma unroll
      for (int r = 0; r < 4; ++r) {
        r0[r] = *(const float4*)(b0r + (size_t)r * FDIM);
        r1[r] = *(const float4*)(b1r + (size_t)r * FDIM);
      }
#pragma unroll
      for (int c = 0; c < 4; ++c) {
        uint2 p0, p1;
        p0.x = pack2bf(f4c(r0[0], c), f4c(r0[1], c));
        p0.y = pack2bf(f4c(r0[2], c), f4c(r0[3], c));
        p1.x = pack2bf(f4c(r1[0], c), f4c(r1[1], c));
        p1.y = pack2bf(f4c(r1[2], c), f4c(r1[3], c));
        *(uint2*)&ldsB0[(bnn + c) * BK + bkq] = p0;
        *(uint2*)&ldsB1[(bnn + c) * BK + bkq] = p1;
      }
      if (kk + BK < DDIM) {   // global_prefetch_b8 of next weight tile
        __builtin_prefetch(b0r + (size_t)BK * FDIM, 0, 1);
        __builtin_prefetch(b1r + (size_t)BK * FDIM, 0, 1);
      }
    }
    __syncthreads();

    // ---- fragments ----
    FragU afr[4];
#pragma unroll
    for (int i = 0; i < 4; ++i) {
      const unsigned short* rp = &ldsA[(wm * 64 + i * 16 + lm) * BK];
      afr[i].u[0] = *(const uint4*)(rp + 8 * hi);        // K = 8h .. 8h+7
      afr[i].u[1] = *(const uint4*)(rp + 16 + 8 * hi);   // K = 16+8h ..
    }
    FragU bfr0[2], bfr1[2];
#pragma unroll
    for (int j = 0; j < 2; ++j) {
      const unsigned short* np0 = &ldsB0[(wn * 32 + j * 16 + lm) * BK];
      const unsigned short* np1 = &ldsB1[(wn * 32 + j * 16 + lm) * BK];
      bfr0[j].u[0] = ((const uint4*)np0)[hi];
      bfr0[j].u[1] = ((const uint4*)np0)[2 + hi];
      bfr1[j].u[0] = ((const uint4*)np1)[hi];
      bfr1[j].u[1] = ((const uint4*)np1)[2 + hi];
    }
    // ---- 16 WMMAs per wave per k-step ----
#pragma unroll
    for (int i = 0; i < 4; ++i)
#pragma unroll
      for (int j = 0; j < 2; ++j) {
        acc0[i][j] = __builtin_amdgcn_wmma_f32_16x16x32_bf16(
            false, afr[i].v, false, bfr0[j].v, (short)0, acc0[i][j], false, false);
        acc1[i][j] = __builtin_amdgcn_wmma_f32_16x16x32_bf16(
            false, afr[i].v, false, bfr1[j].v, (short)0, acc1[i][j], false, false);
      }
    __syncthreads();
  }

  // ---- fused SiLU * gate epilogue, bf16 store ----
#pragma unroll
  for (int i = 0; i < 4; ++i)
#pragma unroll
    for (int j = 0; j < 2; ++j) {
      int col = bn0 + wn * 32 + j * 16 + lm;
#pragma unroll
      for (int r = 0; r < 8; ++r) {
        int row = bm0 + wm * 64 + i * 16 + r + 8 * hi;
        float a = acc0[i][j][r];
        float b = acc1[i][j][r];
        float hv = (a / (1.f + __expf(-a))) * b;   // silu(a) * b
        Hb[(size_t)row * FDIM + col] = f2bf(hv);
      }
    }
}

// ---------------------------------------------------------------------------
// Stage 2 per expert: out (+)= diag(we[:,e]) * (Hb @ woe). Same tiling.
// A tile is pure bf16 copy -> async global->LDS path (ASYNCcnt).
// ---------------------------------------------------------------------------
__global__ __launch_bounds__(256, 1)
void moe_out_kernel(const unsigned short* __restrict__ Hb,  // [N][F] bf16
                    const float* __restrict__ wo,           // [F][D] this expert
                    const float* __restrict__ weE,          // &WE[e], stride NEXP
                    float* __restrict__ out,
                    int accumulate) {
  __shared__ __align__(16) unsigned short ldsA[BM * BK];  // [m][k]
  __shared__ __align__(16) unsigned short ldsB[BN * BK];  // [n][k]

  const int tid  = threadIdx.x;
  const int lane = tid & 31;
  const int wv   = tid >> 5;
  const int wm   = wv >> 2;
  const int wn   = wv & 3;
  const int bn0  = blockIdx.x * BN;
  const int bm0  = blockIdx.y * BM;

  const int ar  = tid >> 1;
  const int ah  = (tid & 1) * 16;
  const int bkq = (tid >> 5) * 4;
  const int bnn = (tid & 31) * 4;

  v8f acc[4][2];
#pragma unroll
  for (int i = 0; i < 4; ++i)
#pragma unroll
    for (int j = 0; j < 2; ++j)
#pragma unroll
      for (int r = 0; r < 8; ++r) acc[i][j][r] = 0.f;

  const int hi = lane >> 4;
  const int lm = lane & 15;

  for (int kk = 0; kk < FDIM; kk += BK) {
    // A tile: Hb already bf16 -> async global->LDS copy (32 B/thread)
    copy32_g2l_async(Hb + (size_t)(bm0 + ar) * FDIM + kk + ah,
                     &ldsA[ar * BK + ah]);
    {   // B tile: wo fp32 -> bf16 transposed, b64 stores
      const float* br = wo + (size_t)(kk + bkq) * DDIM + bn0 + bnn;
      float4 r0[4];
#pragma unroll
      for (int r = 0; r < 4; ++r) r0[r] = *(const float4*)(br + (size_t)r * DDIM);
#pragma unroll
      for (int c = 0; c < 4; ++c) {
        uint2 p;
        p.x = pack2bf(f4c(r0[0], c), f4c(r0[1], c));
        p.y = pack2bf(f4c(r0[2], c), f4c(r0[3], c));
        *(uint2*)&ldsB[(bnn + c) * BK + bkq] = p;
      }
      if (kk + BK < FDIM)
        __builtin_prefetch(br + (size_t)BK * DDIM, 0, 1);
    }
    wait_async0();
    __syncthreads();

    FragU afr[4];
#pragma unroll
    for (int i = 0; i < 4; ++i) {
      const unsigned short* rp = &ldsA[(wm * 64 + i * 16 + lm) * BK];
      afr[i].u[0] = *(const uint4*)(rp + 8 * hi);
      afr[i].u[1] = *(const uint4*)(rp + 16 + 8 * hi);
    }
    FragU bfr[2];
#pragma unroll
    for (int j = 0; j < 2; ++j) {
      const unsigned short* np = &ldsB[(wn * 32 + j * 16 + lm) * BK];
      bfr[j].u[0] = ((const uint4*)np)[hi];
      bfr[j].u[1] = ((const uint4*)np)[2 + hi];
    }
#pragma unroll
    for (int i = 0; i < 4; ++i)
#pragma unroll
      for (int j = 0; j < 2; ++j)
        acc[i][j] = __builtin_amdgcn_wmma_f32_16x16x32_bf16(
            false, afr[i].v, false, bfr[j].v, (short)0, acc[i][j], false, false);
    __syncthreads();
  }

  // epilogue: scale by routing weight, write or accumulate
#pragma unroll
  for (int i = 0; i < 4; ++i) {
    float wr[8];
#pragma unroll
    for (int r = 0; r < 8; ++r) {
      int row = bm0 + wm * 64 + i * 16 + r + 8 * hi;
      wr[r] = weE[(size_t)row * NEXP];
    }
#pragma unroll
    for (int j = 0; j < 2; ++j) {
      int col = bn0 + wn * 32 + j * 16 + lm;
#pragma unroll
      for (int r = 0; r < 8; ++r) {
        int row = bm0 + wm * 64 + i * 16 + r + 8 * hi;
        float v = wr[r] * acc[i][j][r];
        size_t idx = (size_t)row * DDIM + col;
        float prev = accumulate ? out[idx] : 0.0f;
        out[idx] = prev + v;
      }
    }
  }
}

// ---------------------------------------------------------------------------
extern "C" void kernel_launch(void* const* d_in, const int* in_sizes, int n_in,
                              void* d_out, int out_size, void* d_ws, size_t ws_size,
                              hipStream_t stream) {
  (void)in_sizes; (void)n_in; (void)out_size; (void)ws_size;
  const float* x  = (const float*)d_in[0];   // [B,S,D]
  const float* wg = (const float*)d_in[1];   // [D,E]
  const float* w0 = (const float*)d_in[2];   // [E,D,F]
  const float* w1 = (const float*)d_in[3];   // [E,D,F]
  const float* wo = (const float*)d_in[4];   // [E,F,D]
  float* out = (float*)d_out;                // [B,S,D]

  char* ws = (char*)d_ws;
  unsigned short* Hb = (unsigned short*)ws;                       // N*F bf16 (64 MB)
  float* WE = (float*)(ws + (size_t)N_TOK * FDIM * 2);            // N*E fp32

  // gating: 8 tokens (waves) per 256-thread block
  gate_kernel<<<N_TOK / 8, 256, 0, stream>>>(x, wg, WE);

  dim3 g1(FDIM / BN, N_TOK / BM);   // 32 x 64
  dim3 g2(DDIM / BN, N_TOK / BM);   //  8 x 64
  for (int e = 0; e < NEXP; ++e) {
    const float* w0e = w0 + (size_t)e * DDIM * FDIM;
    const float* w1e = w1 + (size_t)e * DDIM * FDIM;
    const float* woe = wo + (size_t)e * FDIM * DDIM;
    moe_h_kernel<<<g1, 256, 0, stream>>>(x, w0e, w1e, Hb);
    moe_out_kernel<<<g2, 256, 0, stream>>>(Hb, woe, WE + e, out, e > 0 ? 1 : 0);
  }
}
